// GoSpecificWattiPooling_82935818485977
// MI455X (gfx1250) — matrix operand order repaired and checked by reading, not compiled, not run on hardware
//
#include <hip/hip_runtime.h>
#include <hip/hip_bf16.h>
#include <math.h>

// ---------------------------------------------------------------------------
// GoSpecificWattiPooling for MI455X (gfx1250, wave32, WMMA)
//   K = H Wk ; Q = G Wq ; S = Q K^T * scale ; alpha = softmax_L(mask(S))
//   Z = alpha H
// f16 staging + v_wmma_f32_16x16x32_f16; intermediates stay hot in 192MB L2.
// Wave tile 32Mx64N (8 accumulators), block tile 64Mx256N (8 waves, 2Mx4N).
// ---------------------------------------------------------------------------

typedef __attribute__((ext_vector_type(16))) _Float16 v16h;
typedef __attribute__((ext_vector_type(8)))  _Float16 v8h;
typedef __attribute__((ext_vector_type(4)))  _Float16 v4h;
typedef __attribute__((ext_vector_type(8)))  float    v8f;

#define ATT_SCALE 0.0625f   // 256^-0.5

// ----------------------------- fragment loads ------------------------------
// A operand: 16 (M rows) x 32 (K) f16, contiguous K per lane (matches B):
//   lanes 0-15 : row M=lane,    K = [0..15]
//   lanes 16-31: row M=lane-16, K = [16..31]
// pa = per-lane base: A + (lane&15)*lda + ((lane>>4)<<4)
__device__ __forceinline__ v16h load_fragA(const _Float16* __restrict__ pa, int k0) {
    return *reinterpret_cast<const v16h*>(pa + k0);
}

// B operand (hardware B is KxN; we store row-major Bmat[N][K] -> C = A*Bmat^T)
// lane indexes N; one contiguous 32-byte run of K per lane:
//   lanes 0-15 : row N=lane,    K = [k0..k0+15]
//   lanes 16-31: row N=lane-16, K = [k0+16..k0+31]
// pb = per-lane base: B + (lane&15)*ldb + ((lane>>4)<<4)
__device__ __forceinline__ v16h load_fragB(const _Float16* __restrict__ pb, int k0) {
    return *reinterpret_cast<const v16h*>(pb + k0);
}

// ------------------------- generic batched WMMA GEMM -----------------------
// C[b] = A[b] (MxK, row-major f16) * Bmat[b]^T (Bmat is NxK row-major f16)
// 256 threads = 8 waves arranged 2(M) x 4(N); block tile 64M x 256N;
// each wave: 32M x 64N via 2x4 f32 accumulators (8 WMMA per 6 loads).
template <typename OutT>
__global__ __launch_bounds__(256) void gemm_att(
    const _Float16* __restrict__ A, const _Float16* __restrict__ Bm,
    OutT* __restrict__ C, int lda, int ldb, int ldc, int K,
    long long sA, long long sB, long long sC)
{
    const int lane = threadIdx.x & 31;
    const int wave = threadIdx.x >> 5;
    const int wm = wave & 1;   // 0..1 -> 32-row M sub-tile
    const int wn = wave >> 1;  // 0..3 -> 64-col N sub-tile

    const long long b = blockIdx.z;
    const _Float16* Ab = A + b * sA + (long long)(blockIdx.y * 64 + wm * 32) * lda;
    const _Float16* Bb = Bm + b * sB + (long long)(blockIdx.x * 256 + wn * 64) * ldb;
    OutT* Cb = C + b * sC;

    // per-lane streaming pointers
    const _Float16* pa0 = Ab + (long long)(lane & 15) * lda + ((lane >> 4) << 4);
    const _Float16* pa1 = pa0 + (long long)16 * lda;
    const _Float16* pb0 = Bb + (long long)(lane & 15) * ldb + ((lane >> 4) << 4);
    const _Float16* pb1 = pb0 + (long long)16 * ldb;
    const _Float16* pb2 = pb0 + (long long)32 * ldb;
    const _Float16* pb3 = pb0 + (long long)48 * ldb;

    v8f acc[2][4] = {};

    for (int k = 0; k < K; k += 32) {
        const v16h a0 = load_fragA(pa0, k);
        const v16h a1 = load_fragA(pa1, k);
        const v16h b0 = load_fragB(pb0, k);
        const v16h b1 = load_fragB(pb1, k);
        const v16h b2 = load_fragB(pb2, k);
        const v16h b3 = load_fragB(pb3, k);
        acc[0][0] = __builtin_amdgcn_wmma_f32_16x16x32_f16(false, a0, false, b0, (short)0, acc[0][0], false, false);
        acc[0][1] = __builtin_amdgcn_wmma_f32_16x16x32_f16(false, a0, false, b1, (short)0, acc[0][1], false, false);
        acc[0][2] = __builtin_amdgcn_wmma_f32_16x16x32_f16(false, a0, false, b2, (short)0, acc[0][2], false, false);
        acc[0][3] = __builtin_amdgcn_wmma_f32_16x16x32_f16(false, a0, false, b3, (short)0, acc[0][3], false, false);
        acc[1][0] = __builtin_amdgcn_wmma_f32_16x16x32_f16(false, a1, false, b0, (short)0, acc[1][0], false, false);
        acc[1][1] = __builtin_amdgcn_wmma_f32_16x16x32_f16(false, a1, false, b1, (short)0, acc[1][1], false, false);
        acc[1][2] = __builtin_amdgcn_wmma_f32_16x16x32_f16(false, a1, false, b2, (short)0, acc[1][2], false, false);
        acc[1][3] = __builtin_amdgcn_wmma_f32_16x16x32_f16(false, a1, false, b3, (short)0, acc[1][3], false, false);
        // hide HBM latency on the streaming operands (emits global_prefetch_b8)
        __builtin_prefetch(pa0 + k + 256, 0, 3);
        __builtin_prefetch(pb0 + k + 256, 0, 3);
    }

    // C/D layout: VGPR j -> M = j + 8*(lane>=16), N = lane&15
    const int cn0 = blockIdx.x * 256 + wn * 64 + (lane & 15);
    const int cm0 = blockIdx.y * 64 + wm * 32 + ((lane >> 4) << 3);
#pragma unroll
    for (int ms = 0; ms < 2; ++ms) {
#pragma unroll
        for (int j = 0; j < 8; ++j) {
            const long long r = (long long)(cm0 + ms * 16 + j) * ldc;
            Cb[r + cn0     ] = (OutT)acc[ms][0][j];
            Cb[r + cn0 + 16] = (OutT)acc[ms][1][j];
            Cb[r + cn0 + 32] = (OutT)acc[ms][2][j];
            Cb[r + cn0 + 48] = (OutT)acc[ms][3][j];
        }
    }
}

// ------------------- fp32 -> f16 cast (+32x32 LDS transpose) ---------------
// src: fp32 [R x Cc] (per batch z). dstT: f16 [Cc x R]. dstRow: optional f16 [R x Cc].
__global__ __launch_bounds__(256) void cast_transpose(
    const float* __restrict__ src, _Float16* __restrict__ dstT,
    _Float16* __restrict__ dstRow, int R, int Cc)
{
    __shared__ float tile[32][33];
    const long long z = blockIdx.z;
    src += z * (long long)R * Cc;
    dstT += z * (long long)R * Cc;
    if (dstRow) dstRow += z * (long long)R * Cc;

    const int c0 = blockIdx.x * 32, r0 = blockIdx.y * 32;
    const int tx = threadIdx.x, ty = threadIdx.y;
#pragma unroll
    for (int i = 0; i < 4; ++i) {
        const int r = r0 + ty + 8 * i;
        const float v = src[(long long)r * Cc + c0 + tx];
        tile[ty + 8 * i][tx] = v;
        if (dstRow) dstRow[(long long)r * Cc + c0 + tx] = (_Float16)v;
    }
    __syncthreads();
#pragma unroll
    for (int i = 0; i < 4; ++i) {
        const int c = c0 + ty + 8 * i;
        dstT[(long long)c * R + r0 + tx] = (_Float16)tile[tx][ty + 8 * i];
    }
}

// vectorized cast: 4 elements/thread (b128 in, b64 out)
__global__ void cast_f32_f16_v4(const float* __restrict__ src,
                                _Float16* __restrict__ dst, long long n4)
{
    const long long i = blockIdx.x * 256LL + threadIdx.x;
    if (i < n4) {
        const float4 v = reinterpret_cast<const float4*>(src)[i];
        v4h o; o[0] = (_Float16)v.x; o[1] = (_Float16)v.y;
        o[2] = (_Float16)v.z; o[3] = (_Float16)v.w;
        reinterpret_cast<v4h*>(dst)[i] = o;
    }
}

// --------------------------- masked softmax over L -------------------------
__device__ __forceinline__ float block_red(float v, bool do_max, float* red)
{
    const int lane = threadIdx.x & 31;
    const int wave = threadIdx.x >> 5;
#pragma unroll
    for (int off = 16; off > 0; off >>= 1) {
        const float o = __shfl_xor(v, off, 32);
        v = do_max ? fmaxf(v, o) : (v + o);
    }
    if (lane == 0) red[wave] = v;
    __syncthreads();
    float r = red[0];
#pragma unroll
    for (int i = 1; i < 8; ++i) r = do_max ? fmaxf(r, red[i]) : (r + red[i]);
    __syncthreads();
    return r;
}

// one 256-thread block per (b,t) row; L = 2048 -> 8 elements/thread (2x float4)
__global__ __launch_bounds__(256) void softmax_rows(
    const float* __restrict__ S, const int* __restrict__ mask,
    _Float16* __restrict__ alpha, int T, int L)
{
    __shared__ float red[8];
    const int row = blockIdx.x;            // b*T + t
    const int b = row / T;
    const float4* s4 = reinterpret_cast<const float4*>(S + (long long)row * L);
    const int4*   m4 = reinterpret_cast<const int4*>(mask + (long long)b * L);
    v4h*          a4 = reinterpret_cast<v4h*>(alpha + (long long)row * L);
    const int tid = threadIdx.x;

    float vals[8];
    float mx = -INFINITY;
#pragma unroll
    for (int c = 0; c < 2; ++c) {
        const float4 v = s4[tid + c * 256];
        const int4   m = m4[tid + c * 256];
        vals[4 * c + 0] = (m.x != 0) ? -INFINITY : v.x * ATT_SCALE;
        vals[4 * c + 1] = (m.y != 0) ? -INFINITY : v.y * ATT_SCALE;
        vals[4 * c + 2] = (m.z != 0) ? -INFINITY : v.z * ATT_SCALE;
        vals[4 * c + 3] = (m.w != 0) ? -INFINITY : v.w * ATT_SCALE;
    }
#pragma unroll
    for (int i = 0; i < 8; ++i) mx = fmaxf(mx, vals[i]);
    mx = block_red(mx, true, red);

    float sum = 0.f;
#pragma unroll
    for (int i = 0; i < 8; ++i) {
        const float e = __expf(vals[i] - mx);   // exp(-inf)=0 for masked
        vals[i] = e;
        sum += e;
    }
    sum = block_red(sum, false, red);

    const float inv = 1.f / sum;
#pragma unroll
    for (int c = 0; c < 2; ++c) {
        v4h o;
#pragma unroll
        for (int j = 0; j < 4; ++j) o[j] = (_Float16)(vals[4 * c + j] * inv);
        a4[tid + c * 256] = o;
    }
}

// ------------------------------- host side ---------------------------------
extern "C" void kernel_launch(void* const* d_in, const int* in_sizes, int n_in,
                              void* d_out, int out_size, void* d_ws, size_t ws_size,
                              hipStream_t stream)
{
    (void)in_sizes; (void)n_in; (void)out_size; (void)ws_size;
    const int B = 8, L = 2048, T = 512, Dh = 1024, Dg = 768, Dp = 256;

    const float* H    = (const float*)d_in[0];   // [B,L,Dh]
    const float* G    = (const float*)d_in[1];   // [B,T,Dg]
    const int*   mask = (const int*)  d_in[2];   // [B,L]
    const float* Wk   = (const float*)d_in[3];   // [Dh,Dp]
    const float* Wq   = (const float*)d_in[4];   // [Dg,Dp]
    float* Z = (float*)d_out;                    // [B,T,Dh]

    char* ws = (char*)d_ws;
    size_t off = 0;
    auto carve = [&](size_t bytes) -> char* {
        char* p = ws + off;
        off += (bytes + 255) & ~(size_t)255;
        return p;
    };
    _Float16* Hf16  = (_Float16*)carve((size_t)B * L * Dh * 2);  // [B*L, Dh]
    _Float16* Ht    = (_Float16*)carve((size_t)B * Dh * L * 2);  // [B][Dh, L]
    _Float16* Gf16  = (_Float16*)carve((size_t)B * T * Dg * 2);  // [B*T, Dg]
    _Float16* Wkt   = (_Float16*)carve((size_t)Dp * Dh * 2);     // [Dp, Dh]
    _Float16* Wqt   = (_Float16*)carve((size_t)Dp * Dg * 2);     // [Dp, Dg]
    _Float16* Kf16  = (_Float16*)carve((size_t)B * L * Dp * 2);  // [B][L, Dp]
    _Float16* Qf16  = (_Float16*)carve((size_t)B * T * Dp * 2);  // [B][T, Dp]
    float*    Sbuf  = (float*)   carve((size_t)B * T * L * 4);   // [B][T, L]
    _Float16* Abuf  = (_Float16*)carve((size_t)B * T * L * 2);   // [B][T, L]

    const dim3 tb(32, 8);

    // Stage 1: f16 staging. H -> Hf16 (row-major) + Ht (transposed per batch).
    cast_transpose<<<dim3(Dh / 32, L / 32, B), tb, 0, stream>>>(H, Ht, Hf16, L, Dh);
    // Wk [Dh,Dp] -> Wkt [Dp,Dh]; Wq [Dg,Dp] -> Wqt [Dp,Dg]
    cast_transpose<<<dim3(Dp / 32, Dh / 32, 1), tb, 0, stream>>>(Wk, Wkt, nullptr, Dh, Dp);
    cast_transpose<<<dim3(Dp / 32, Dg / 32, 1), tb, 0, stream>>>(Wq, Wqt, nullptr, Dg, Dp);
    cast_f32_f16_v4<<<((B * T * Dg / 4) + 255) / 256, 256, 0, stream>>>(
        G, Gf16, (long long)B * T * Dg / 4);

    // Stage 2: projections (C = A * Bmat^T), outputs f16.
    //   K = Hf16 * Wkt^T : M=B*L, N=Dp, K=Dh
    gemm_att<_Float16><<<dim3(Dp / 256, (B * L) / 64, 1), 256, 0, stream>>>(
        Hf16, Wkt, Kf16, Dh, Dh, Dp, Dh, 0, 0, 0);
    //   Q = Gf16 * Wqt^T : M=B*T, N=Dp, K=Dg
    gemm_att<_Float16><<<dim3(Dp / 256, (B * T) / 64, 1), 256, 0, stream>>>(
        Gf16, Wqt, Qf16, Dg, Dg, Dp, Dg, 0, 0, 0);

    // Stage 3: logits S[b] = Q[b] * K[b]^T : M=T, N=L, K=Dp (batched, f32 out)
    gemm_att<float><<<dim3(L / 256, T / 64, B), 256, 0, stream>>>(
        Qf16, Kf16, Sbuf, Dp, Dp, L, Dp,
        (long long)T * Dp, (long long)L * Dp, (long long)T * L);

    // Stage 4: masked + scaled softmax over L, alpha stored f16
    softmax_rows<<<B * T, 256, 0, stream>>>(Sbuf, mask, Abuf, T, L);

    // Stage 5: Z[b] = alpha[b] * (Ht[b])^T : M=T, N=Dh, K=L (batched, f32 out)
    gemm_att<float><<<dim3(Dh / 256, T / 64, B), 256, 0, stream>>>(
        Abuf, Ht, Z, L, L, Dh, L,
        (long long)T * L, (long long)Dh * L, (long long)T * Dh);
}
